// CausalSelfAttention_76759655514352
// MI455X (gfx1250) — matrix operand order repaired and checked by reading, not compiled, not run on hardware
//
#include <hip/hip_runtime.h>

typedef __attribute__((ext_vector_type(16))) _Float16 v16h;
typedef __attribute__((ext_vector_type(8)))  float    v8f;

#define T_SEQ  2048
#define C_DIM  2048
#define QKVD   3072
#define NHEAD  32
#define NGRP   8
#define HD     64

// ---------- CDNA5 async global->LDS copy (ASYNCcnt-tracked) ----------
__device__ __forceinline__ void async_ld_b128(unsigned int lds_byte_off, const void* gptr) {
  asm volatile("global_load_async_to_lds_b128 %0, %1, off"
               :: "v"(lds_byte_off), "v"((unsigned long long)(size_t)gptr)
               : "memory");
}
__device__ __forceinline__ void wait_async0() {
  asm volatile("s_wait_asynccnt 0x0" ::: "memory");
}

// ---------- WMMA fragment loaders (wave32, 16x16x32 f16) ----------
// A (16xK=32) layout: lane L(0-15)=row M=L, halves j:
//   K = hi*16 + j (contiguous, matches B), hi = lane>>4
__device__ __forceinline__ int ka_off(int j, int hi) {
  return hi * 16 + j;
}

__device__ __forceinline__ v16h load_a_rm(const _Float16* A, int m0, int k0, int ld, int lane) {
  const _Float16* p = A + (size_t)(m0 + (lane & 15)) * ld + k0;
  const int hi = lane >> 4;
  v16h a;
#pragma unroll
  for (int j = 0; j < 16; ++j) a[j] = p[ka_off(j, hi)];
  return a;
}

// B (K=32 x 16): lanes 0-15 hold K=0..15, lanes 16-31 hold K=16..31 (contiguous)
// from N-major storage W[N][K] (B[k][n] = W[n0+n][k0+k]) -> fully contiguous 32B/lane
__device__ __forceinline__ v16h load_b_nk(const _Float16* W, int n0, int k0, int ld, int lane) {
  const _Float16* p = W + (size_t)(n0 + (lane & 15)) * ld + k0 + (lane >> 4) * 16;
  v16h b;
#pragma unroll
  for (int j = 0; j < 16; ++j) b[j] = p[j];
  return b;
}

// ---------- fp32 -> fp16 conversion ----------
__global__ void __launch_bounds__(256) f32_to_f16_kernel(const float* __restrict__ src,
                                                         _Float16* __restrict__ dst, int n) {
  int i = blockIdx.x * blockDim.x + threadIdx.x;
  if (i < n) dst[i] = (_Float16)src[i];
}

// ---------- LDS-staged GEMM: C[m,n] = sum_k A[m,k]*B[n,k] + bias[n] ----------
// Block tile 128(M) x 128(N); 8 waves as 4(M) x 2(N); wave tile 32x64.
// K staged in 32-wide slices, double-buffered in LDS via async global->LDS.
#define GEMM_BM 128
#define GEMM_BN 128
#define GEMM_BK 32

__global__ void __launch_bounds__(256) gemm_nt_bias(const _Float16* __restrict__ A,
                                                    const _Float16* __restrict__ B,
                                                    const float* __restrict__ bias,
                                                    float* __restrict__ C,
                                                    int M, int N, int K) {
  __shared__ _Float16 ldsA[2][GEMM_BM * GEMM_BK];  // 2 x 8 KiB
  __shared__ _Float16 ldsB[2][GEMM_BN * GEMM_BK];  // 2 x 8 KiB
  const int t    = threadIdx.x;
  const int lane = t & 31;
  const int wv   = t >> 5;
  const int mw   = wv >> 1;          // 0..3
  const int nw   = wv & 1;           // 0..1
  const int mblk = blockIdx.y * GEMM_BM;
  const int nblk = blockIdx.x * GEMM_BN;

  // Stage one 128x32 slice of A and B: 512 chunks of 16B each, 2 per thread.
  auto stage = [&](int buf, int k0) {
    const unsigned int la = (unsigned int)(size_t)(&ldsA[buf][0]);
    const unsigned int lb = (unsigned int)(size_t)(&ldsB[buf][0]);
#pragma unroll
    for (int i = 0; i < 2; ++i) {
      const int c   = t + i * 256;       // 0..511
      const int row = c >> 2;            // 0..127
      const int seg = c & 3;             // 16B segment within 64B row
      async_ld_b128(la + (unsigned int)(row * 64 + seg * 16),
                    A + (size_t)(mblk + row) * K + k0 + seg * 8);
      async_ld_b128(lb + (unsigned int)(row * 64 + seg * 16),
                    B + (size_t)(nblk + row) * K + k0 + seg * 8);
    }
  };

  v8f acc[2][4] = {};

  stage(0, 0);
  wait_async0();
  __syncthreads();

  int buf = 0;
  for (int k0 = 0; k0 < K; k0 += GEMM_BK) {
    if (k0 + GEMM_BK < K) stage(buf ^ 1, k0 + GEMM_BK);  // prefetch next slice

    const _Float16* tA = &ldsA[buf][0];
    const _Float16* tB = &ldsB[buf][0];
    v16h a0 = load_a_rm(tA, mw * 32,      0, GEMM_BK, lane);
    v16h a1 = load_a_rm(tA, mw * 32 + 16, 0, GEMM_BK, lane);
#pragma unroll
    for (int j = 0; j < 4; ++j) {
      v16h b = load_b_nk(tB, nw * 64 + j * 16, 0, GEMM_BK, lane);
      acc[0][j] = __builtin_amdgcn_wmma_f32_16x16x32_f16(false, a0, false, b,
                                                         (short)0, acc[0][j], false, false);
      acc[1][j] = __builtin_amdgcn_wmma_f32_16x16x32_f16(false, a1, false, b,
                                                         (short)0, acc[1][j], false, false);
    }

    wait_async0();      // prefetch landed
    __syncthreads();    // everyone done reading buf, next slice visible
    buf ^= 1;
  }

  const int cb = lane & 15, hi = lane >> 4;
#pragma unroll
  for (int ms = 0; ms < 2; ++ms) {
#pragma unroll
    for (int j = 0; j < 4; ++j) {
      const int cc = nblk + nw * 64 + j * 16 + cb;
      const float bv = bias ? bias[cc] : 0.0f;
#pragma unroll
      for (int r = 0; r < 8; ++r) {
        const int row = mblk + mw * 32 + ms * 16 + hi * 8 + r;
        C[(size_t)row * N + cc] = acc[ms][j][r] + bv;
      }
    }
  }
}

// ---------- RoPE + head split ----------
// qkv fp32 [T][3072] -> Q[32][T][64], K[8][T][64], Vt[8][64][T]  (all f16)
// V is stored TRANSPOSED so the PV B-fragments in attention are contiguous.
__global__ void __launch_bounds__(256) rope_split(const float* __restrict__ qkv,
                                                  const float* __restrict__ cosb,
                                                  const float* __restrict__ sinb,
                                                  _Float16* __restrict__ Qm,
                                                  _Float16* __restrict__ Km,
                                                  _Float16* __restrict__ Vt) {
  int idx = blockIdx.x * blockDim.x + threadIdx.x;
  if (idx >= T_SEQ * QKVD) return;
  const int t = idx / QKVD;
  const int j = idx - t * QKVD;
  const int g = j / 384;
  const int r = (j - g * 384) / 64;
  const int d = j & 63;
  float v = qkv[idx];
  const bool isq = (r < 4);
  const bool isk = (r == 4);
  if ((isq || isk) && d < 16) {
    const float c = cosb[t * 16 + d];
    const float s = sinb[t * 16 + d];
    const float partner = qkv[(size_t)t * QKVD + g * 384 + r * 64 + ((d < 8) ? d + 8 : d - 8)];
    v = (d < 8) ? (v * c - partner * s) : (v * c + partner * s);
  }
  const _Float16 h = (_Float16)v;
  if (isq)      Qm[((size_t)(g * 4 + r) * T_SEQ + t) * HD + d] = h;
  else if (isk) Km[((size_t)g * T_SEQ + t) * HD + d] = h;
  else          Vt[((size_t)g * HD + d) * T_SEQ + t] = h;   // transposed: [g][dim][token]
}

// ---------- Flash-style causal attention: one wave per 16-row query tile ----------
__global__ void __launch_bounds__(256) attn_fwd(const _Float16* __restrict__ Qm,
                                                const _Float16* __restrict__ Km,
                                                const _Float16* __restrict__ Vt,
                                                _Float16* __restrict__ Ym) {
  __shared__ _Float16 plds[8][16 * 32];  // per-wave P transpose staging (1KB each)
  const int lane = threadIdx.x & 31;
  const int wv   = threadIdx.x >> 5;
  const int head = blockIdx.y;
  const int grp  = head >> 2;  // Q_PER_KV = 4
  const int q0   = (blockIdx.x * 8 + wv) * 16;
  const _Float16* Qh  = Qm + (size_t)head * T_SEQ * HD;
  const _Float16* Kh  = Km + (size_t)grp  * T_SEQ * HD;
  const _Float16* Vth = Vt + (size_t)grp  * HD * T_SEQ;     // [dim][token]
  const int cb = lane & 15, hi = lane >> 4;

  // Q fragments, pre-scaled by 1/sqrt(64)
  v16h qa0 = load_a_rm(Qh, q0, 0,  HD, lane);
  v16h qa1 = load_a_rm(Qh, q0, 32, HD, lane);
#pragma unroll
  for (int j = 0; j < 16; ++j) {
    qa0[j] = qa0[j] * (_Float16)0.125f;
    qa1[j] = qa1[j] * (_Float16)0.125f;
  }

  v8f o[4] = {};
  float mrow[8], lrow[8];
#pragma unroll
  for (int r = 0; r < 8; ++r) { mrow[r] = -1e30f; lrow[r] = 0.0f; }

  const int kb_max = (q0 + 15) >> 5;
  for (int kb = 0; kb <= kb_max; ++kb) {
    const int kt0 = kb * 32;
    // S tile 16x32 = Q(16x64) . K^T(64x32), two column sub-tiles
    v8f s0 = {}, s1 = {};
    {
      v16h b = load_b_nk(Kh, kt0, 0, HD, lane);
      s0 = __builtin_amdgcn_wmma_f32_16x16x32_f16(false, qa0, false, b, (short)0, s0, false, false);
      b = load_b_nk(Kh, kt0, 32, HD, lane);
      s0 = __builtin_amdgcn_wmma_f32_16x16x32_f16(false, qa1, false, b, (short)0, s0, false, false);
      b = load_b_nk(Kh, kt0 + 16, 0, HD, lane);
      s1 = __builtin_amdgcn_wmma_f32_16x16x32_f16(false, qa0, false, b, (short)0, s1, false, false);
      b = load_b_nk(Kh, kt0 + 16, 32, HD, lane);
      s1 = __builtin_amdgcn_wmma_f32_16x16x32_f16(false, qa1, false, b, (short)0, s1, false, false);
    }

    // causal mask (only needed near the diagonal)
    if (kt0 + 31 > q0) {
#pragma unroll
      for (int r = 0; r < 8; ++r) {
        const int rowtok = q0 + hi * 8 + r;
        if (kt0 + cb      > rowtok) s0[r] = -1e30f;
        if (kt0 + 16 + cb > rowtok) s1[r] = -1e30f;
      }
    }

    // online softmax: row stats live across 16 lanes of a half-wave
    float p0[8], p1[8];
#pragma unroll
    for (int r = 0; r < 8; ++r) {
      float mx = fmaxf(s0[r], s1[r]);
#pragma unroll
      for (int off = 1; off < 16; off <<= 1) mx = fmaxf(mx, __shfl_xor(mx, off, 32));
      const float mn = fmaxf(mrow[r], mx);
      const float al = __expf(mrow[r] - mn);
      mrow[r] = mn;
      p0[r] = __expf(s0[r] - mn);
      p1[r] = __expf(s1[r] - mn);
      float rs = p0[r] + p1[r];
#pragma unroll
      for (int off = 1; off < 16; off <<= 1) rs += __shfl_xor(rs, off, 32);
      lrow[r] = lrow[r] * al + rs;
#pragma unroll
      for (int j = 0; j < 4; ++j) o[j][r] = o[j][r] * al;
    }

    // transpose P (C-layout -> A-layout) through per-wave LDS
    _Float16* pt = plds[wv];
#pragma unroll
    for (int r = 0; r < 8; ++r) {
      pt[(hi * 8 + r) * 32 + cb]      = (_Float16)p0[r];
      pt[(hi * 8 + r) * 32 + cb + 16] = (_Float16)p1[r];
    }
    asm volatile("s_wait_dscnt 0" ::: "memory");  // wave-private tile: DS-count wait suffices
    v16h pa = load_a_rm(pt, 0, 0, 32, lane);

    // O(16x64) += P(16x32) . V(32x64); B from transposed V -> contiguous loads
#pragma unroll
    for (int j = 0; j < 4; ++j) {
      v16h vb = load_b_nk(Vth, j * 16, kt0, T_SEQ, lane);  // B[k][n] = Vt[n][k]
      o[j] = __builtin_amdgcn_wmma_f32_16x16x32_f16(false, pa, false, vb, (short)0, o[j], false, false);
    }
  }

  // epilogue: normalize and store y[t][head*64 + d] as f16
#pragma unroll
  for (int j = 0; j < 4; ++j) {
#pragma unroll
    for (int r = 0; r < 8; ++r) {
      const int row = q0 + hi * 8 + r;
      Ym[(size_t)row * (NHEAD * HD) + head * HD + j * 16 + cb] = (_Float16)(o[j][r] / lrow[r]);
    }
  }
}

// ---------- host-side launcher ----------
extern "C" void kernel_launch(void* const* d_in, const int* in_sizes, int n_in,
                              void* d_out, int out_size, void* d_ws, size_t ws_size,
                              hipStream_t stream) {
  const float* x      = (const float*)d_in[0];
  const float* cosb   = (const float*)d_in[1];
  const float* sinb   = (const float*)d_in[2];
  const float* W_attn = (const float*)d_in[3];
  const float* b_attn = (const float*)d_in[4];
  const float* W_proj = (const float*)d_in[5];
  const float* b_proj = (const float*)d_in[6];

  char* ws = (char*)d_ws;
  // workspace layout (bytes)
  _Float16* xh   = (_Float16*)(ws + 0);                    //  8.0 MiB  x  f16 [T][C]
  _Float16* wah  = (_Float16*)(ws + 8388608);              // 12.0 MiB  W_attn f16 [3072][2048]
  _Float16* wph  = (_Float16*)(ws + 20971520);             //  8.0 MiB  W_proj f16 [2048][2048]
  float*    qkvf = (float*)   (ws + 29360128);             // 24.0 MiB  qkv fp32 [T][3072]
  _Float16* Qh   = (_Float16*)(ws + 54525952);             //  8.0 MiB  Q f16 [32][T][64]
  _Float16* Kh   = (_Float16*)(ws + 62914560);             //  2.0 MiB  K f16 [8][T][64]
  _Float16* Vth  = (_Float16*)(ws + 65011712);             //  2.0 MiB  V f16 [8][64][T] (transposed)
  _Float16* Yh   = (_Float16*)(ws + 67108864);             //  8.0 MiB  y f16 [T][2048]

  // 1) downconvert inputs to f16
  {
    int n = T_SEQ * C_DIM;
    f32_to_f16_kernel<<<(n + 255) / 256, 256, 0, stream>>>(x, xh, n);
    n = QKVD * C_DIM;
    f32_to_f16_kernel<<<(n + 255) / 256, 256, 0, stream>>>(W_attn, wah, n);
    n = C_DIM * (NHEAD * HD);
    f32_to_f16_kernel<<<(n + 255) / 256, 256, 0, stream>>>(W_proj, wph, n);
  }

  // 2) QKV projection: qkv = x @ W_attn^T + b_attn   (M=2048, N=3072, K=2048)
  {
    dim3 grid(QKVD / GEMM_BN, T_SEQ / GEMM_BM);  // 24 x 16
    gemm_nt_bias<<<grid, 256, 0, stream>>>(xh, wah, b_attn, qkvf, T_SEQ, QKVD, C_DIM);
  }

  // 3) RoPE + split into per-head Q/K/Vt (f16; V transposed)
  {
    const int n = T_SEQ * QKVD;
    rope_split<<<(n + 255) / 256, 256, 0, stream>>>(qkvf, cosb, sinb, Qh, Kh, Vth);
  }

  // 4) causal flash attention -> y f16 [T][2048]
  {
    dim3 grid(T_SEQ / 128, NHEAD);  // 8 waves x 16 rows = 128 query rows per block
    attn_fwd<<<grid, 256, 0, stream>>>(Qh, Kh, Vth, Yh);
  }

  // 5) output projection: out = y @ W_proj^T + b_proj   (M=2048, N=2048, K=2048)
  {
    dim3 grid(C_DIM / GEMM_BN, T_SEQ / GEMM_BM);  // 16 x 16
    gemm_nt_bias<<<grid, 256, 0, stream>>>(Yh, wph, b_proj, (float*)d_out, T_SEQ, C_DIM, C_DIM);
  }
}
